// SelfAttention_67568425500919
// MI455X (gfx1250) — matrix operand order repaired and checked by reading, not compiled, not run on hardware
//
#include <hip/hip_runtime.h>
#include <hip/hip_bf16.h>

typedef __attribute__((ext_vector_type(16))) _Float16 v16h;
typedef __attribute__((ext_vector_type(8)))  _Float16 v8h;
typedef __attribute__((ext_vector_type(8)))  float    v8f;

#define BATCH 4
#define SEQ   2048
#define DIM   512

union HFrag { v16h v; v8h h[2]; };

// D = A x B + C, 16x16x32 f16 -> f32
__device__ __forceinline__ v8f wmma16(v16h a, v16h b, v8f c) {
    return __builtin_amdgcn_wmma_f32_16x16x32_f16(
        /*neg_a=*/false, a, /*neg_b=*/false, b,
        /*c_mod=*/(short)0, c, /*reuse_a=*/false, /*reuse_b=*/false);
}

__device__ __forceinline__ v8f zero8() {
    return (v8f){0.f, 0.f, 0.f, 0.f, 0.f, 0.f, 0.f, 0.f};
}

// A fragment (16x32, f16) from a row-major matrix with leading dim ld.
// Row M = lane%16; lanes 0-15 hold K = kk+{0..7,16..23},
// lanes 16-31 hold K = kk+{8..15,24..31} (v16h elems 0..7 / 8..15).
__device__ __forceinline__ v16h load_a_rowmajor(const _Float16* base, int ld,
                                                int kk, int lane) {
    int  m   = lane & 15;
    int  off = (lane < 16) ? 0 : 8;
    const _Float16* p = base + (size_t)m * ld + kk + off;
    HFrag a;
    a.h[0] = *(const v8h*)(p);       // K = kk+off   .. +7
    a.h[1] = *(const v8h*)(p + 16);  // K = kk+16+off.. +7
    return a.v;
}

// B fragment (32x16, f16) from an "N-major" matrix: row n holds K contiguously
// (leading dim ld). Col N = lane%16; lanes 0-15: K=kk+{0..7,16..23},
// lanes 16-31: K=kk+{8..15,24..31}.
__device__ __forceinline__ v16h load_b_kcontig(const _Float16* base, int ld,
                                               int kk, int lane) {
    int  n   = lane & 15;
    int  off = (lane < 16) ? 0 : 8;
    const _Float16* p = base + (size_t)n * ld + kk + off;
    HFrag b;
    b.h[0] = *(const v8h*)(p);
    b.h[1] = *(const v8h*)(p + 16);
    return b.v;
}

// Build A fragment (16x32) from a row of f32 probabilities in LDS.
__device__ __forceinline__ v16h build_a_from_lds(const float* sS, int t,
                                                 int n, int lane) {
    const float* p0 = sS + n * SEQ + t + ((lane < 16) ? 0 : 8);
    float fa[16];
    *(float4*)&fa[0]  = *(const float4*)(p0);
    *(float4*)&fa[4]  = *(const float4*)(p0 + 4);
    *(float4*)&fa[8]  = *(const float4*)(p0 + 16);
    *(float4*)&fa[12] = *(const float4*)(p0 + 20);
    v16h a;
#pragma unroll
    for (int i = 0; i < 16; ++i) a[i] = (_Float16)fa[i];
    return a;
}

// ---------------- conversion kernels ----------------

__global__ void cvt_x_kernel(const float* __restrict__ X,
                             _Float16* __restrict__ Xh, int n) {
    int i = blockIdx.x * 256 + threadIdx.x;
    if (i < n) Xh[i] = (_Float16)X[i];
}

// W[k][n] f32 -> Wt[z][n][k] f16 (transposed so K is contiguous per out col)
__global__ void cvt_w_kernel(const float* __restrict__ Wq,
                             const float* __restrict__ Wk,
                             const float* __restrict__ Wv,
                             _Float16* __restrict__ WtAll) {
    int z = blockIdx.y;
    const float* W = (z == 0) ? Wq : (z == 1) ? Wk : Wv;
    int i = blockIdx.x * 256 + threadIdx.x;     // i < DIM*DIM
    int k = i >> 9;          // / DIM
    int n = i & (DIM - 1);   // % DIM
    WtAll[(size_t)z * DIM * DIM + (size_t)n * DIM + k] = (_Float16)W[i];
}

// ---------------- QKV projection ----------------
// grid (8192/16, 1, 3), block 256 (8 waves).
// Each wave: one 16x64 output slice = 4 accumulators sharing one A fragment
// per K-step. Ping-pong double buffering: two named fragment sets alternate
// (no rotation copies); each WMMA group runs with the other set's loads in
// flight.
__global__ void qkv_kernel(const _Float16* __restrict__ Xh,
                           const _Float16* __restrict__ WtAll,
                           const float* __restrict__ bq,
                           const float* __restrict__ bk,
                           const float* __restrict__ bv,
                           _Float16* __restrict__ Qh,
                           _Float16* __restrict__ Kh,
                           _Float16* __restrict__ Vt) {
    int z    = blockIdx.z;
    int wave = threadIdx.x >> 5;
    int lane = threadIdx.x & 31;

    const _Float16* Wt   = WtAll + (size_t)z * DIM * DIM;
    const float*    bias = (z == 0) ? bq : (z == 1) ? bk : bv;

    int rowG    = blockIdx.x * 16;   // global row (b*SEQ + t)
    int colBase = wave * 64;         // output feature base (16x64 slice)

    const _Float16* abase = Xh + (size_t)rowG * DIM;
    const _Float16* wbase = Wt + (size_t)colBase * DIM;

    v8f c[4];
#pragma unroll
    for (int i = 0; i < 4; ++i) c[i] = zero8();

    // Set A <- step 0
    v16h aA  = load_a_rowmajor(abase, DIM, 0, lane);
    v16h pA0 = load_b_kcontig(wbase + (size_t)0  * DIM, DIM, 0, lane);
    v16h pA1 = load_b_kcontig(wbase + (size_t)16 * DIM, DIM, 0, lane);
    v16h pA2 = load_b_kcontig(wbase + (size_t)32 * DIM, DIM, 0, lane);
    v16h pA3 = load_b_kcontig(wbase + (size_t)48 * DIM, DIM, 0, lane);
    v16h aB, pB0, pB1, pB2, pB3;

#pragma unroll 1
    for (int kk = 0; kk < DIM - 64; kk += 64) {
        // Set B <- step kk+32
        aB  = load_a_rowmajor(abase, DIM, kk + 32, lane);
        pB0 = load_b_kcontig(wbase + (size_t)0  * DIM, DIM, kk + 32, lane);
        pB1 = load_b_kcontig(wbase + (size_t)16 * DIM, DIM, kk + 32, lane);
        pB2 = load_b_kcontig(wbase + (size_t)32 * DIM, DIM, kk + 32, lane);
        pB3 = load_b_kcontig(wbase + (size_t)48 * DIM, DIM, kk + 32, lane);
        // Compute step kk with set A
        c[0] = wmma16(aA, pA0, c[0]);
        c[1] = wmma16(aA, pA1, c[1]);
        c[2] = wmma16(aA, pA2, c[2]);
        c[3] = wmma16(aA, pA3, c[3]);
        // Set A <- step kk+64
        aA  = load_a_rowmajor(abase, DIM, kk + 64, lane);
        pA0 = load_b_kcontig(wbase + (size_t)0  * DIM, DIM, kk + 64, lane);
        pA1 = load_b_kcontig(wbase + (size_t)16 * DIM, DIM, kk + 64, lane);
        pA2 = load_b_kcontig(wbase + (size_t)32 * DIM, DIM, kk + 64, lane);
        pA3 = load_b_kcontig(wbase + (size_t)48 * DIM, DIM, kk + 64, lane);
        // Compute step kk+32 with set B
        c[0] = wmma16(aB, pB0, c[0]);
        c[1] = wmma16(aB, pB1, c[1]);
        c[2] = wmma16(aB, pB2, c[2]);
        c[3] = wmma16(aB, pB3, c[3]);
    }
    // Epilogue: set A holds DIM-64; load set B <- DIM-32.
    aB  = load_a_rowmajor(abase, DIM, DIM - 32, lane);
    pB0 = load_b_kcontig(wbase + (size_t)0  * DIM, DIM, DIM - 32, lane);
    pB1 = load_b_kcontig(wbase + (size_t)16 * DIM, DIM, DIM - 32, lane);
    pB2 = load_b_kcontig(wbase + (size_t)32 * DIM, DIM, DIM - 32, lane);
    pB3 = load_b_kcontig(wbase + (size_t)48 * DIM, DIM, DIM - 32, lane);
    c[0] = wmma16(aA, pA0, c[0]);
    c[1] = wmma16(aA, pA1, c[1]);
    c[2] = wmma16(aA, pA2, c[2]);
    c[3] = wmma16(aA, pA3, c[3]);
    c[0] = wmma16(aB, pB0, c[0]);
    c[1] = wmma16(aB, pB1, c[1]);
    c[2] = wmma16(aB, pB2, c[2]);
    c[3] = wmma16(aB, pB3, c[3]);

    int n    = lane & 15;
    int off8 = (lane < 16) ? 0 : 8;
    int bidx   = rowG / SEQ;
    int tLocal = rowG % SEQ;

#pragma unroll
    for (int c4 = 0; c4 < 4; ++c4) {
        int   col = colBase + c4 * 16 + n;
        float bb  = bias[col];
        if (z == 0) {
#pragma unroll
            for (int r = 0; r < 8; ++r)
                Qh[(size_t)(rowG + r + off8) * DIM + col] =
                    (_Float16)(c[c4][r] + bb);
        } else if (z == 1) {
#pragma unroll
            for (int r = 0; r < 8; ++r)
                Kh[(size_t)(rowG + r + off8) * DIM + col] =
                    (_Float16)(c[c4][r] + bb);
        } else {
#pragma unroll
            for (int r = 0; r < 8; ++r)
                Vt[(size_t)bidx * DIM * SEQ + (size_t)col * SEQ +
                   (tLocal + r + off8)] = (_Float16)(c[c4][r] + bb);
        }
    }
}

// ---------------- fused attention ----------------
// grid (SEQ/16, BATCH), block 256 (8 waves).
// LDS: scores[16][SEQ] f32 (128 KB) + reduction scratch.
#define SMEM_FLOATS (16 * SEQ + 256 + 16 + 16)

__global__ void attn_kernel(const _Float16* __restrict__ Qh,
                            const _Float16* __restrict__ Kh,
                            const _Float16* __restrict__ Vt,
                            float* __restrict__ Out) {
    extern __shared__ float smem[];
    float* sS     = smem;                // 16 * SEQ
    float* red    = smem + 16 * SEQ;     // 256
    float* rowsum = red + 256;           // 16
    float* rowmax = rowsum + 16;         // 16

    int b     = blockIdx.y;
    int qbase = blockIdx.x * 16;
    int wave  = threadIdx.x >> 5;
    int lane  = threadIdx.x & 31;
    int n     = lane & 15;
    int off8  = (lane < 16) ? 0 : 8;

    const float scale = 0.04419417382415922f;  // 1/sqrt(512)
    const _Float16* qrows = Qh + ((size_t)b * SEQ + qbase) * DIM;

    // ---- Phase 1: scores = (Q K^T) * scale, into LDS.
    // 2 key tiles per pass share one A fragment; ping-pong pipelined K loop.
    for (int j = 0; j < 8; ++j) {
        int tbase0 = (wave * 2 + j * 16) * 16;
        int tbase1 = tbase0 + 16;
        const _Float16* krows0 = Kh + ((size_t)b * SEQ + tbase0) * DIM;
        const _Float16* krows1 = Kh + ((size_t)b * SEQ + tbase1) * DIM;
        v8f c0 = zero8(), c1 = zero8();

        v16h aA = load_a_rowmajor(qrows, DIM, 0, lane);
        v16h pA0 = load_b_kcontig(krows0, DIM, 0, lane);
        v16h pA1 = load_b_kcontig(krows1, DIM, 0, lane);
        v16h aB, pB0, pB1;
#pragma unroll 1
        for (int kk = 0; kk < DIM - 64; kk += 64) {
            aB  = load_a_rowmajor(qrows, DIM, kk + 32, lane);
            pB0 = load_b_kcontig(krows0, DIM, kk + 32, lane);
            pB1 = load_b_kcontig(krows1, DIM, kk + 32, lane);
            c0 = wmma16(aA, pA0, c0);
            c1 = wmma16(aA, pA1, c1);
            aA  = load_a_rowmajor(qrows, DIM, kk + 64, lane);
            pA0 = load_b_kcontig(krows0, DIM, kk + 64, lane);
            pA1 = load_b_kcontig(krows1, DIM, kk + 64, lane);
            c0 = wmma16(aB, pB0, c0);
            c1 = wmma16(aB, pB1, c1);
        }
        aB  = load_a_rowmajor(qrows, DIM, DIM - 32, lane);
        pB0 = load_b_kcontig(krows0, DIM, DIM - 32, lane);
        pB1 = load_b_kcontig(krows1, DIM, DIM - 32, lane);
        c0 = wmma16(aA, pA0, c0);
        c1 = wmma16(aA, pA1, c1);
        c0 = wmma16(aB, pB0, c0);
        c1 = wmma16(aB, pB1, c1);

#pragma unroll
        for (int r = 0; r < 8; ++r) {
            int m = r + off8;
            sS[m * SEQ + tbase0 + n] = c0[r] * scale;
            sS[m * SEQ + tbase1 + n] = c1[r] * scale;
        }
    }
    __syncthreads();

    // ---- Phase 2: row softmax in LDS (unnormalized exp; row sums kept) ----
    {
        int row = threadIdx.x >> 4;   // 0..15
        int li  = threadIdx.x & 15;   // 0..15
        float* p = sS + row * SEQ + li * (SEQ / 16);
        float mx = -1e30f;
        for (int i = 0; i < SEQ / 16; ++i) mx = fmaxf(mx, p[i]);
        red[row * 16 + li] = mx;
        __syncthreads();
        if (li == 0) {
            float m2 = red[row * 16];
            for (int i = 1; i < 16; ++i) m2 = fmaxf(m2, red[row * 16 + i]);
            rowmax[row] = m2;
        }
        __syncthreads();
        float mrow = rowmax[row];
        float sum = 0.f;
        for (int i = 0; i < SEQ / 16; ++i) {
            float e = __expf(p[i] - mrow);
            p[i] = e;
            sum += e;
        }
        red[row * 16 + li] = sum;
        __syncthreads();
        if (li == 0) {
            float s2 = 0.f;
            for (int i = 0; i < 16; ++i) s2 += red[row * 16 + i];
            rowsum[row] = s2;
        }
        __syncthreads();
    }

    // ---- Phase 3: O = P @ V (each wave owns a 16x64 output slice).
    // Ping-pong pipelined over t; A fragments rebuilt from LDS each step
    // (DS latency is short; the cvt VALU co-executes with the matrix pipe).
    int colBase = wave * 64;
    v8f acc[4];
#pragma unroll
    for (int i = 0; i < 4; ++i) acc[i] = zero8();

    const _Float16* vbase =
        Vt + (size_t)b * DIM * SEQ + (size_t)colBase * SEQ;

    v16h fA0 = load_b_kcontig(vbase + (size_t)0  * SEQ, SEQ, 0, lane);
    v16h fA1 = load_b_kcontig(vbase + (size_t)16 * SEQ, SEQ, 0, lane);
    v16h fA2 = load_b_kcontig(vbase + (size_t)32 * SEQ, SEQ, 0, lane);
    v16h fA3 = load_b_kcontig(vbase + (size_t)48 * SEQ, SEQ, 0, lane);
    v16h fB0, fB1, fB2, fB3;

#pragma unroll 1
    for (int t = 0; t < SEQ - 64; t += 64) {
        fB0 = load_b_kcontig(vbase + (size_t)0  * SEQ, SEQ, t + 32, lane);
        fB1 = load_b_kcontig(vbase + (size_t)16 * SEQ, SEQ, t + 32, lane);
        fB2 = load_b_kcontig(vbase + (size_t)32 * SEQ, SEQ, t + 32, lane);
        fB3 = load_b_kcontig(vbase + (size_t)48 * SEQ, SEQ, t + 32, lane);
        {
            v16h a = build_a_from_lds(sS, t, n, lane);
            acc[0] = wmma16(a, fA0, acc[0]);
            acc[1] = wmma16(a, fA1, acc[1]);
            acc[2] = wmma16(a, fA2, acc[2]);
            acc[3] = wmma16(a, fA3, acc[3]);
        }
        fA0 = load_b_kcontig(vbase + (size_t)0  * SEQ, SEQ, t + 64, lane);
        fA1 = load_b_kcontig(vbase + (size_t)16 * SEQ, SEQ, t + 64, lane);
        fA2 = load_b_kcontig(vbase + (size_t)32 * SEQ, SEQ, t + 64, lane);
        fA3 = load_b_kcontig(vbase + (size_t)48 * SEQ, SEQ, t + 64, lane);
        {
            v16h a = build_a_from_lds(sS, t + 32, n, lane);
            acc[0] = wmma16(a, fB0, acc[0]);
            acc[1] = wmma16(a, fB1, acc[1]);
            acc[2] = wmma16(a, fB2, acc[2]);
            acc[3] = wmma16(a, fB3, acc[3]);
        }
    }
    // Epilogue: set A holds SEQ-64; load set B <- SEQ-32.
    fB0 = load_b_kcontig(vbase + (size_t)0  * SEQ, SEQ, SEQ - 32, lane);
    fB1 = load_b_kcontig(vbase + (size_t)16 * SEQ, SEQ, SEQ - 32, lane);
    fB2 = load_b_kcontig(vbase + (size_t)32 * SEQ, SEQ, SEQ - 32, lane);
    fB3 = load_b_kcontig(vbase + (size_t)48 * SEQ, SEQ, SEQ - 32, lane);
    {
        v16h a = build_a_from_lds(sS, SEQ - 64, n, lane);
        acc[0] = wmma16(a, fA0, acc[0]);
        acc[1] = wmma16(a, fA1, acc[1]);
        acc[2] = wmma16(a, fA2, acc[2]);
        acc[3] = wmma16(a, fA3, acc[3]);
    }
    {
        v16h a = build_a_from_lds(sS, SEQ - 32, n, lane);
        acc[0] = wmma16(a, fB0, acc[0]);
        acc[1] = wmma16(a, fB1, acc[1]);
        acc[2] = wmma16(a, fB2, acc[2]);
        acc[3] = wmma16(a, fB3, acc[3]);
    }

    float invs[8];
#pragma unroll
    for (int r = 0; r < 8; ++r) invs[r] = 1.0f / rowsum[r + off8];

#pragma unroll
    for (int c4 = 0; c4 < 4; ++c4) {
#pragma unroll
        for (int r = 0; r < 8; ++r) {
            int m = r + off8;
            Out[((size_t)b * SEQ + qbase + m) * DIM + colBase + c4 * 16 + n] =
                acc[c4][r] * invs[r];
        }
    }
}

// ---------------- host launcher ----------------
extern "C" void kernel_launch(void* const* d_in, const int* in_sizes, int n_in,
                              void* d_out, int out_size, void* d_ws, size_t ws_size,
                              hipStream_t stream) {
    const float* X  = (const float*)d_in[0];
    const float* Wq = (const float*)d_in[1];
    const float* bq = (const float*)d_in[2];
    const float* Wk = (const float*)d_in[3];
    const float* bk = (const float*)d_in[4];
    const float* Wv = (const float*)d_in[5];
    const float* bv = (const float*)d_in[6];
    float* Out = (float*)d_out;

    char* ws = (char*)d_ws;
    const size_t nX  = (size_t)BATCH * SEQ * DIM;     // 4,194,304
    const size_t nW  = (size_t)DIM * DIM;             // 262,144
    _Float16* Xh    = (_Float16*)(ws);
    _Float16* WtAll = (_Float16*)(ws + nX * 2);
    _Float16* Qh    = (_Float16*)(ws + nX * 2 + 3 * nW * 2);
    _Float16* Kh    = (_Float16*)(ws + nX * 2 + 3 * nW * 2 + nX * 2);
    _Float16* Vt    = (_Float16*)(ws + nX * 2 + 3 * nW * 2 + 2 * nX * 2);

    cvt_x_kernel<<<(int)(nX / 256), 256, 0, stream>>>(X, Xh, (int)nX);
    cvt_w_kernel<<<dim3((unsigned)(nW / 256), 3), 256, 0, stream>>>(Wq, Wk, Wv, WtAll);

    qkv_kernel<<<dim3(BATCH * SEQ / 16, 1, 3), 256, 0, stream>>>(
        Xh, WtAll, bq, bk, bv, Qh, Kh, Vt);

    attn_kernel<<<dim3(SEQ / 16, BATCH), 256, SMEM_FLOATS * sizeof(float),
                  stream>>>(Qh, Kh, Vt, Out);
}
